// PositionAwareLayer_49495203119136
// MI455X (gfx1250) — matrix operand order repaired and checked by reading, not compiled, not run on hardware
//
#include <hip/hip_runtime.h>

typedef __attribute__((ext_vector_type(16))) _Float16 v16h;
typedef __attribute__((ext_vector_type(8)))  float    v8f;

#define HDIM 128
#define HQ   32

// ---------- helpers ----------
__device__ __forceinline__ unsigned pk2(float a, float b) {
    union { _Float16 h[2]; unsigned u; } t;
    t.h[0] = (_Float16)a; t.h[1] = (_Float16)b;
    return t.u;
}
__device__ __forceinline__ unsigned short f2h(float a) {
    union { _Float16 h; unsigned short u; } t;
    t.h = (_Float16)a;
    return t.u;
}
// low 32 bits of a generic pointer into LDS == wave-relative LDS byte address
__device__ __forceinline__ unsigned ldsAddr(const void* p) {
    return (unsigned)(unsigned long long)p;
}

// async global->LDS 16-byte copy (CDNA5 Tensor/Async path, ASYNCcnt tracked)
#define ASYNC_CP16(ldsoff, gptr, imm)                                          \
    asm volatile("global_load_async_to_lds_b128 %0, %1, off offset:%2"         \
                 :: "v"(ldsoff), "v"((const void*)(gptr)), "n"(imm) : "memory")

// ---------- zero workspace (agg accumulator) ----------
__global__ __launch_bounds__(256)
void pal_zero_kernel(float4* p, size_t n4) {
    size_t i = (size_t)blockIdx.x * blockDim.x + threadIdx.x;
    size_t st = (size_t)gridDim.x * blockDim.x;
    float4 z; z.x = 0.f; z.y = 0.f; z.z = 0.f; z.w = 0.f;
    for (; i < n4; i += st) p[i] = z;
}

// ---------- convert x [N,128] fp32 -> f16 once (saves ~32x redundant cvt in gather) ----------
__global__ __launch_bounds__(256)
void pal_xconv_kernel(const float2* __restrict__ xin, unsigned* __restrict__ xo, size_t n2) {
    size_t i = (size_t)blockIdx.x * blockDim.x + threadIdx.x;
    size_t st = (size_t)gridDim.x * blockDim.x;
    for (; i < n2; i += st) {
        float2 f = xin[i];
        xo[i] = pk2(f.x, f.y);
    }
}

// ---------- convert fp32 weights [Ktot x Nout] into WMMA B-fragment f16 order ----------
// B layout (16-bit, 32x16): lane L: N = L%16; element e(0..15): K = (e/8)*16 + (L/16)*8 + e%8
__global__ __launch_bounds__(256)
void pal_wconv_kernel(const float* __restrict__ W, unsigned short* __restrict__ F,
                      int Ktot, int Nout) {
    int nt16 = Nout >> 4;
    int total = (Ktot >> 5) * nt16 * 512;
    for (int t = blockIdx.x * blockDim.x + threadIdx.x; t < total;
         t += gridDim.x * blockDim.x) {
        int frag = t >> 9, rem = t & 511;
        int lane = rem >> 4, e = rem & 15;
        int kc = frag / nt16, nt = frag - kc * nt16;
        int K = kc * 32 + ((e >> 3) << 4) + ((lane >> 4) << 3) + (e & 7);
        int Nn = nt * 16 + (lane & 15);
        F[t] = f2h(W[(size_t)K * Nout + Nn]);
    }
}

// ---------- pos MLP: [E,3] -> relu(@pw1 3x32 +b) @pw2 32x32 +b -> f16 [E,32] ----------
__global__ __launch_bounds__(256)
void pal_posfeat_kernel(const float* __restrict__ pos, const long long* __restrict__ ei,
                        const float* __restrict__ pw1, const float* __restrict__ pb1,
                        const float* __restrict__ pw2, const float* __restrict__ pb2,
                        unsigned short* __restrict__ pf, int E) {
    int e = blockIdx.x * blockDim.x + threadIdx.x;
    if (e >= E) return;
    long long r = ei[e], c = ei[E + e];
    float d0 = pos[r * 3 + 0] - pos[c * 3 + 0];
    float d1 = pos[r * 3 + 1] - pos[c * 3 + 1];
    float d2 = pos[r * 3 + 2] - pos[c * 3 + 2];
    float h[32];
#pragma unroll
    for (int j = 0; j < 32; ++j) {
        float a = d0 * pw1[j] + d1 * pw1[32 + j] + d2 * pw1[64 + j] + pb1[j];
        h[j] = a > 0.f ? a : 0.f;
    }
    unsigned short* o = pf + (size_t)e * HQ;
#pragma unroll 4
    for (int j = 0; j < 32; ++j) {
        float a = pb2[j];
#pragma unroll
        for (int i = 0; i < 32; ++i) a += h[i] * pw2[i * 32 + j];
        o[j] = f2h(a);
    }
}

// LDS geometry (in halves)
#define W1H   36864   // 9 kc * 8 nt * 512
#define W2H   16384   // 4 kc * 8 nt * 512
#define EWTOT (W1H + W2H)           // 53248 halves = 106496 B
#define ESTR  296                   // msg row stride (halves), 16B aligned, conflict-free
#define ESTG  (16 * ESTR)           // 4736 halves per wave
#define HSTR  136                   // hidden row stride (halves)

// ---------- edge message kernel: per-edge 288->128->128 MLP + atomic scatter ----------
__global__ __launch_bounds__(128)
void pal_edge_kernel(const unsigned short* __restrict__ xh, const long long* __restrict__ ei,
                     const unsigned short* __restrict__ pf,
                     const uint4* __restrict__ wfrag,   // mw1 frags | mw2 frags (f16)
                     const float* __restrict__ b1, const float* __restrict__ b2,
                     float* __restrict__ agg, int E, int nTiles) {
    extern __shared__ unsigned short lds[];
    const int wave = threadIdx.x >> 5;
    const int lane = threadIdx.x & 31;
    const int h16 = lane >> 4, l16 = lane & 15;

    // stage weight fragments (106496 B) into LDS once per workgroup
    {
        uint4* dq = (uint4*)lds;
        const int nq = EWTOT / 8;   // 6656 uint4
        for (int i = threadIdx.x; i < nq; i += blockDim.x) dq[i] = wfrag[i];
    }
    __syncthreads();

    const uint4* bq1 = (const uint4*)lds;            // mw1 frags
    const uint4* bq2 = (const uint4*)(lds + W1H);    // mw2 frags
    unsigned short* ms = lds + EWTOT + wave * ESTG;  // per-wave staging tile
    int* scol = (int*)(lds + EWTOT + 4 * ESTG);      // 4 x 16 ints

    float bias1[8], bias2[8];
#pragma unroll
    for (int nt = 0; nt < 8; ++nt) {
        bias1[nt] = b1[nt * 16 + l16];
        bias2[nt] = b2[nt * 16 + l16];
    }

    const int totalWaves = gridDim.x * 4;
    for (int tile = blockIdx.x * 4 + wave; tile < nTiles; tile += totalWaves) {
        const int e0 = tile * 16;
        // ---- async-stage msg tile: [xh[row] | xh[col] | posfeat], f16 rows of 288 ----
        {
            int r = lane >> 1, p = lane & 1;
            long long nr = ei[e0 + r];
            long long nc = ei[E + e0 + r];
            if (p == 0) scol[wave * 16 + r] = (int)nc;
            unsigned d0 = ldsAddr(ms + r * ESTR + p * 64);
            unsigned d1 = ldsAddr(ms + r * ESTR + 128 + p * 64);
            const unsigned short* g0 = xh + (size_t)nr * HDIM + p * 64;
            const unsigned short* g1 = xh + (size_t)nc * HDIM + p * 64;
            ASYNC_CP16(d0, g0, 0);   ASYNC_CP16(d1, g1, 0);
            ASYNC_CP16(d0, g0, 16);  ASYNC_CP16(d1, g1, 16);
            ASYNC_CP16(d0, g0, 32);  ASYNC_CP16(d1, g1, 32);
            ASYNC_CP16(d0, g0, 48);  ASYNC_CP16(d1, g1, 48);
            ASYNC_CP16(d0, g0, 64);  ASYNC_CP16(d1, g1, 64);
            ASYNC_CP16(d0, g0, 80);  ASYNC_CP16(d1, g1, 80);
            ASYNC_CP16(d0, g0, 96);  ASYNC_CP16(d1, g1, 96);
            ASYNC_CP16(d0, g0, 112); ASYNC_CP16(d1, g1, 112);
            unsigned dp = ldsAddr(ms + r * ESTR + 256 + p * 16);
            const unsigned short* gp = pf + (size_t)(e0 + r) * HQ + p * 16;
            ASYNC_CP16(dp, gp, 0);
            ASYNC_CP16(dp, gp, 16);
        }
        asm volatile("s_wait_asynccnt 0" ::: "memory");
        asm volatile("s_wait_dscnt 0" ::: "memory");

        // ---- GEMM1: (16x288) @ (288x128), bias preloaded into accumulators ----
        v8f acc[8];
#pragma unroll
        for (int nt = 0; nt < 8; ++nt)
#pragma unroll
            for (int i = 0; i < 8; ++i) acc[nt][i] = bias1[nt];
#pragma unroll 1
        for (int kc = 0; kc < 9; ++kc) {
            union { v16h v; uint4 q[2]; } A;
            const unsigned short* ab = ms + l16 * ESTR + kc * 32 + h16 * 8;
            A.q[0] = *(const uint4*)ab;
            A.q[1] = *(const uint4*)(ab + 16);
#pragma unroll
            for (int nt = 0; nt < 8; ++nt) {
                union { v16h v; uint4 q[2]; } B;
                const uint4* bp = bq1 + ((kc * 8 + nt) * 32 + lane) * 2;
                B.q[0] = bp[0]; B.q[1] = bp[1];
                acc[nt] = __builtin_amdgcn_wmma_f32_16x16x32_f16(
                    false, A.v, false, B.v, (short)0, acc[nt], false, false);
            }
        }
        // ---- ReLU -> f16 hidden tile in LDS (re-fragment as A for GEMM2) ----
#pragma unroll
        for (int nt = 0; nt < 8; ++nt)
#pragma unroll
            for (int v = 0; v < 8; ++v) {
                float f = acc[nt][v];
                f = f > 0.f ? f : 0.f;
                ms[(v + 8 * h16) * HSTR + nt * 16 + l16] = f2h(f);
            }
        asm volatile("s_wait_dscnt 0" ::: "memory");

        // ---- GEMM2: (16x128) @ (128x128) + atomic scatter into agg[col] ----
        union { v16h v; uint4 q[2]; } A2[4];
#pragma unroll
        for (int kc = 0; kc < 4; ++kc) {
            const unsigned short* ab = ms + l16 * HSTR + kc * 32 + h16 * 8;
            A2[kc].q[0] = *(const uint4*)ab;
            A2[kc].q[1] = *(const uint4*)(ab + 16);
        }
        int cm[8];
#pragma unroll
        for (int v = 0; v < 8; ++v) cm[v] = scol[wave * 16 + v + 8 * h16];
#pragma unroll 1
        for (int nt = 0; nt < 8; ++nt) {
            v8f a2;
#pragma unroll
            for (int i = 0; i < 8; ++i) a2[i] = bias2[nt];
#pragma unroll
            for (int kc = 0; kc < 4; ++kc) {
                union { v16h v; uint4 q[2]; } B;
                const uint4* bp = bq2 + ((kc * 8 + nt) * 32 + lane) * 2;
                B.q[0] = bp[0]; B.q[1] = bp[1];
                a2 = __builtin_amdgcn_wmma_f32_16x16x32_f16(
                    false, A2[kc].v, false, B.v, (short)0, a2, false, false);
            }
#pragma unroll
            for (int v = 0; v < 8; ++v)
                atomicAdd(agg + (size_t)cm[v] * HDIM + nt * 16 + l16, a2[v]);
        }
    }
}

#define U1H   32768   // 8 kc * 8 nt * 512
#define U2H   16384
#define UWTOT (U1H + U2H)   // 49152 halves = 98304 B
#define NSTR  264           // node staging row stride (halves)
#define NSTG  (16 * NSTR)

// ---------- node update kernel: [x|agg] 256->128->128 ----------
__global__ __launch_bounds__(128)
void pal_node_kernel(const unsigned short* __restrict__ xh, const float* __restrict__ agg,
                     const uint4* __restrict__ wfrag,   // uw1 frags | uw2 frags
                     const float* __restrict__ b1, const float* __restrict__ b2,
                     float* __restrict__ out, int nTiles) {
    extern __shared__ unsigned short lds[];
    const int wave = threadIdx.x >> 5;
    const int lane = threadIdx.x & 31;
    const int h16 = lane >> 4, l16 = lane & 15;

    {
        uint4* dq = (uint4*)lds;
        const int nq = UWTOT / 8;
        for (int i = threadIdx.x; i < nq; i += blockDim.x) dq[i] = wfrag[i];
    }
    __syncthreads();

    const uint4* bq1 = (const uint4*)lds;
    const uint4* bq2 = (const uint4*)(lds + U1H);
    unsigned short* ms = lds + UWTOT + wave * NSTG;

    float bias1[8], bias2[8];
#pragma unroll
    for (int nt = 0; nt < 8; ++nt) {
        bias1[nt] = b1[nt * 16 + l16];
        bias2[nt] = b2[nt * 16 + l16];
    }

    const int totalWaves = gridDim.x * 4;
    for (int tile = blockIdx.x * 4 + wave; tile < nTiles; tile += totalWaves) {
        const int n0 = tile * 16;
        // ---- stage [x | agg] tile: x half async f16 copy, agg half cvt ----
        {
            int r = lane >> 1, p = lane & 1;
            unsigned d0 = ldsAddr(ms + r * NSTR + p * 64);
            const unsigned short* g0 = xh + (size_t)(n0 + r) * HDIM + p * 64;
            ASYNC_CP16(d0, g0, 0);
            ASYNC_CP16(d0, g0, 16);
            ASYNC_CP16(d0, g0, 32);
            ASYNC_CP16(d0, g0, 48);
            ASYNC_CP16(d0, g0, 64);
            ASYNC_CP16(d0, g0, 80);
            ASYNC_CP16(d0, g0, 96);
            ASYNC_CP16(d0, g0, 112);
            const float4* s1 = (const float4*)(agg + (size_t)(n0 + r) * HDIM + p * 64);
            uint4* d1 = (uint4*)(ms + r * NSTR + 128 + p * 64);
#pragma unroll
            for (int i = 0; i < 8; ++i) {
                float4 fa = s1[2 * i], fb = s1[2 * i + 1];
                uint4 u;
                u.x = pk2(fa.x, fa.y); u.y = pk2(fa.z, fa.w);
                u.z = pk2(fb.x, fb.y); u.w = pk2(fb.z, fb.w);
                d1[i] = u;
            }
        }
        asm volatile("s_wait_asynccnt 0" ::: "memory");
        asm volatile("s_wait_dscnt 0" ::: "memory");

        // ---- GEMM1: (16x256) @ (256x128) ----
        v8f acc[8];
#pragma unroll
        for (int nt = 0; nt < 8; ++nt)
#pragma unroll
            for (int i = 0; i < 8; ++i) acc[nt][i] = bias1[nt];
#pragma unroll 1
        for (int kc = 0; kc < 8; ++kc) {
            union { v16h v; uint4 q[2]; } A;
            const unsigned short* ab = ms + l16 * NSTR + kc * 32 + h16 * 8;
            A.q[0] = *(const uint4*)ab;
            A.q[1] = *(const uint4*)(ab + 16);
#pragma unroll
            for (int nt = 0; nt < 8; ++nt) {
                union { v16h v; uint4 q[2]; } B;
                const uint4* bp = bq1 + ((kc * 8 + nt) * 32 + lane) * 2;
                B.q[0] = bp[0]; B.q[1] = bp[1];
                acc[nt] = __builtin_amdgcn_wmma_f32_16x16x32_f16(
                    false, A.v, false, B.v, (short)0, acc[nt], false, false);
            }
        }
#pragma unroll
        for (int nt = 0; nt < 8; ++nt)
#pragma unroll
            for (int v = 0; v < 8; ++v) {
                float f = acc[nt][v];
                f = f > 0.f ? f : 0.f;
                ms[(v + 8 * h16) * HSTR + nt * 16 + l16] = f2h(f);
            }
        asm volatile("s_wait_dscnt 0" ::: "memory");

        // ---- GEMM2: (16x128) @ (128x128), direct store ----
        union { v16h v; uint4 q[2]; } A2[4];
#pragma unroll
        for (int kc = 0; kc < 4; ++kc) {
            const unsigned short* ab = ms + l16 * HSTR + kc * 32 + h16 * 8;
            A2[kc].q[0] = *(const uint4*)ab;
            A2[kc].q[1] = *(const uint4*)(ab + 16);
        }
#pragma unroll 1
        for (int nt = 0; nt < 8; ++nt) {
            v8f a2;
#pragma unroll
            for (int i = 0; i < 8; ++i) a2[i] = bias2[nt];
#pragma unroll
            for (int kc = 0; kc < 4; ++kc) {
                union { v16h v; uint4 q[2]; } B;
                const uint4* bp = bq2 + ((kc * 8 + nt) * 32 + lane) * 2;
                B.q[0] = bp[0]; B.q[1] = bp[1];
                a2 = __builtin_amdgcn_wmma_f32_16x16x32_f16(
                    false, A2[kc].v, false, B.v, (short)0, a2, false, false);
            }
#pragma unroll
            for (int v = 0; v < 8; ++v)
                out[(size_t)(n0 + v + 8 * h16) * HDIM + nt * 16 + l16] = a2[v];
        }
    }
}

extern "C" void kernel_launch(void* const* d_in, const int* in_sizes, int n_in,
                              void* d_out, int out_size, void* d_ws, size_t ws_size,
                              hipStream_t stream) {
    const float*     x   = (const float*)d_in[0];
    const float*     pos = (const float*)d_in[1];
    const long long* ei  = (const long long*)d_in[2];
    const float* pw1 = (const float*)d_in[3];
    const float* pb1 = (const float*)d_in[4];
    const float* pw2 = (const float*)d_in[5];
    const float* pb2 = (const float*)d_in[6];
    const float* mw1 = (const float*)d_in[7];
    const float* mb1 = (const float*)d_in[8];
    const float* mw2 = (const float*)d_in[9];
    const float* mb2 = (const float*)d_in[10];
    const float* uw1 = (const float*)d_in[11];
    const float* ub1 = (const float*)d_in[12];
    const float* uw2 = (const float*)d_in[13];
    const float* ub2 = (const float*)d_in[14];
    float* out = (float*)d_out;

    const int N = in_sizes[0] / HDIM;   // 50000
    const int E = in_sizes[2] / 2;      // 800000

    // workspace layout
    char* ws = (char*)d_ws;
    size_t off = 0;
    float* agg = (float*)(ws + off);              off += (size_t)N * HDIM * 4;
    off = (off + 255) & ~(size_t)255;
    unsigned short* xh = (unsigned short*)(ws + off); off += (size_t)N * HDIM * 2;
    off = (off + 255) & ~(size_t)255;
    unsigned short* pf = (unsigned short*)(ws + off); off += (size_t)E * HQ * 2;
    off = (off + 255) & ~(size_t)255;
    unsigned short* mfrag = (unsigned short*)(ws + off); off += (size_t)(W1H + W2H) * 2;
    off = (off + 255) & ~(size_t)255;
    unsigned short* ufrag = (unsigned short*)(ws + off); off += (size_t)(U1H + U2H) * 2;

    pal_zero_kernel<<<2048, 256, 0, stream>>>((float4*)agg, (size_t)N * HDIM / 4);
    pal_xconv_kernel<<<2048, 256, 0, stream>>>((const float2*)x, (unsigned*)xh,
                                               (size_t)N * HDIM / 2);
    pal_wconv_kernel<<<64, 256, 0, stream>>>(mw1, mfrag, 288, 128);
    pal_wconv_kernel<<<32, 256, 0, stream>>>(mw2, mfrag + W1H, 128, 128);
    pal_wconv_kernel<<<64, 256, 0, stream>>>(uw1, ufrag, 256, 128);
    pal_wconv_kernel<<<32, 256, 0, stream>>>(uw2, ufrag + U1H, 128, 128);
    pal_posfeat_kernel<<<(E + 255) / 256, 256, 0, stream>>>(pos, ei, pw1, pb1, pw2, pb2, pf, E);

    const size_t edgeLds = (size_t)(EWTOT + 4 * ESTG) * 2 + 4 * 16 * sizeof(int); // 144640 B
    pal_edge_kernel<<<1024, 128, edgeLds, stream>>>(xh, ei, pf, (const uint4*)mfrag,
                                                    mb1, mb2, agg, E, E / 16);

    const size_t nodeLds = (size_t)(UWTOT + 4 * NSTG) * 2;                        // 132096 B
    pal_node_kernel<<<512, 128, nodeLds, stream>>>(xh, agg, (const uint4*)ufrag,
                                                   ub1, ub2, out, N / 16);
}
